// MyModel2_82652350644592
// MI455X (gfx1250) — matrix operand order repaired and checked by reading, not compiled, not run on hardware
//
#include <hip/hip_runtime.h>
#include <hip/hip_bf16.h>
#include <math.h>

typedef __attribute__((ext_vector_type(16))) __bf16 v16bf;
typedef __attribute__((ext_vector_type(8)))  __bf16 v8bf;
typedef __attribute__((ext_vector_type(8)))  float  v8f;

#define NN 50000

// ---------------- utility kernels ----------------

__global__ void fill_f32(float* p, float v, int n) {
    int i = blockIdx.x * blockDim.x + threadIdx.x;
    if (i < n) p[i] = v;
}

// convert f32 weight (K x Nc, row-major) into bf16 TRANSPOSED (Nc x K) so that
// a WMMA B-fragment (one column of B, contiguous K) is two contiguous 16B runs.
__global__ void f32_to_bf16_T(const float* __restrict__ in, __bf16* __restrict__ out,
                              int K, int Nc) {
    int i = blockIdx.x * blockDim.x + threadIdx.x;
    if (i >= K * Nc) return;
    int k = i / Nc, n = i % Nc;
    out[(size_t)n * K + k] = (__bf16)in[i];
}

// ---------------- pool + layernorm ----------------
__global__ void pool_ln(const float* __restrict__ x, const float* __restrict__ g,
                        const float* __restrict__ bta, __bf16* __restrict__ h) {
    int n = blockIdx.x;
    int t = threadIdx.x;                    // 256 threads, 4 pool outputs each
    const float4* xr = (const float4*)(x + (size_t)n * 2048);
    float vals[4];
    float s = 0.f, ss = 0.f;
#pragma unroll
    for (int i = 0; i < 2; i++) {
        float4 q = xr[t * 2 + i];
        float a = fmaxf(q.x, q.y);
        float b = fmaxf(q.z, q.w);
        vals[2 * i] = a; vals[2 * i + 1] = b;
        s += a + b; ss += a * a + b * b;
    }
    __shared__ float sh0[256];
    __shared__ float sh1[256];
    sh0[t] = s; sh1[t] = ss;
    __syncthreads();
    for (int o = 128; o > 0; o >>= 1) {
        if (t < o) { sh0[t] += sh0[t + o]; sh1[t] += sh1[t + o]; }
        __syncthreads();
    }
    float mu  = sh0[0] * (1.f / 1024.f);
    float var = sh1[0] * (1.f / 1024.f) - mu * mu;
    float rstd = rsqrtf(var + 1e-5f);
#pragma unroll
    for (int i = 0; i < 4; i++) {
        int p = t * 4 + i;
        float hv = (vals[i] - mu) * rstd * g[p] + bta[p];
        h[(size_t)n * 1024 + p] = (__bf16)hv;
    }
}

// ---------------- WMMA bf16 GEMM, LDS-staged B (async DMA) ----------------
// C(M,Nc) = A(M,K)*B + bias, ELU, bf16 out.  BT = B transposed (Nc x K) bf16.
// Block: 256 thr (8 waves) -> 128 rows x 64 cols.  B chunks (64 cols x KC K)
// staged to LDS via global_load_async_to_lds_b64 (ASYNCcnt), double-buffered.
// Requires: K % 64 == 0, Nc % 64 == 0, M % 16 == 0.
#define KC  64
#define BPAD 8
#define BROW (KC + BPAD)          // LDS row stride in elements (144B = 36 banks)

__global__ void gemm_bf16_lds(const __bf16* __restrict__ A, const __bf16* __restrict__ BT,
                              const float* __restrict__ bias,
                              __bf16* __restrict__ outb, int M, int K, int Nc) {
    __shared__ alignas(16) __bf16 ldsB[2][64 * BROW];

    int tid  = threadIdx.x;
    int wave = tid >> 5;
    int lane = tid & 31;
    int lmod = lane & 15;
    int hi   = lane >> 4;

    int mtiles = M >> 4;
    int mt = blockIdx.x * 8 + wave;
    bool active = mt < mtiles;
    if (!active) mt = mtiles - 1;          // clamp: keep EXEC full for barriers/WMMA
    int n0 = blockIdx.y * 64;

    const __bf16* arow = A + (size_t)(mt * 16 + lmod) * K + hi * 8;

    // staging geometry: 64 rows (B columns) x KC elems; 8B units, NT=4 units/thread
    int srow = tid >> 2;                   // 0..63
    int sg   = (tid & 3) << 4;             // 0,16,32,48 : element offset of 32B pair? no:
    // each thread moves 4 units of 8B: units u = tid + j*256 -> row = u>>4, unit g = u&15
    const int nchunks = K / KC;

    auto stage = [&](int c, int buf) {
#pragma unroll
        for (int j = 0; j < 4; j++) {
            int u = tid + j * 256;
            int n = u >> 4;                // B column within tile, 0..63
            int g = (u & 15) << 2;         // element offset (4 elems = 8B per unit)
            unsigned long long gaddr =
                (unsigned long long)(BT + (size_t)(n0 + n) * K + c * KC + g);
            unsigned int laddr =
                (unsigned int)(size_t)&ldsB[buf][n * BROW + g];
            asm volatile("global_load_async_to_lds_b64 %0, %1, off"
                         :: "v"(laddr), "v"(gaddr) : "memory");
        }
    };

    v8f acc[4] = {};

    stage(0, 0);
    asm volatile("s_wait_asynccnt 0x0" ::: "memory");
    __syncthreads();

    for (int c = 0; c < nchunks; c++) {
        int buf = c & 1;
        if (c + 1 < nchunks) stage(c + 1, buf ^ 1);

#pragma unroll
        for (int klocal = 0; klocal < KC; klocal += 32) {
            const __bf16* ak = arow + c * KC + klocal;
            v8bf alo = *(const v8bf*)(ak);
            v8bf ahi = *(const v8bf*)(ak + 16);
            v16bf a = __builtin_shufflevector(alo, ahi, 0, 1, 2, 3, 4, 5, 6, 7,
                                              8, 9, 10, 11, 12, 13, 14, 15);
#pragma unroll
            for (int t = 0; t < 4; t++) {
                const __bf16* bp = &ldsB[buf][(t * 16 + lmod) * BROW + klocal + hi * 8];
                v8bf blo = *(const v8bf*)(bp);       // ds_load_b128
                v8bf bhi = *(const v8bf*)(bp + 16);  // ds_load_b128
                v16bf b = __builtin_shufflevector(blo, bhi, 0, 1, 2, 3, 4, 5, 6, 7,
                                                  8, 9, 10, 11, 12, 13, 14, 15);
                acc[t] = __builtin_amdgcn_wmma_f32_16x16x32_bf16(
                    false, a, false, b, (short)0, acc[t], false, false);
            }
        }
        if (c + 1 < nchunks) {
            asm volatile("s_wait_asynccnt 0x0" ::: "memory");
            __syncthreads();
        }
    }

    if (active) {
        int mrow = mt * 16 + hi * 8;
#pragma unroll
        for (int t = 0; t < 4; t++) {
            int n = n0 + t * 16 + lmod;
            float bv = bias[n];
#pragma unroll
            for (int v = 0; v < 8; v++) {
                float val = acc[t][v] + bv;
                val = val > 0.f ? val : (expf(val) - 1.f);
                outb[(size_t)(mrow + v) * Nc + n] = (__bf16)val;
            }
        }
    }
}

// ---------------- WMMA bf16 GEMM, direct-global (small K/Nc) ----------------
template <int NT>
__global__ void gemm_bf16_small(const __bf16* __restrict__ A, const __bf16* __restrict__ BT,
                                float* __restrict__ outf, int M, int K, int Nc) {
    int wave = threadIdx.x >> 5;
    int lane = threadIdx.x & 31;
    int mt = blockIdx.x * (blockDim.x >> 5) + wave;
    if (mt * 16 >= M) return;              // wave-uniform exit
    int lmod = lane & 15;
    int hi = lane >> 4;

    const __bf16* arow = A + (size_t)(mt * 16 + lmod) * K + hi * 8;
    const __bf16* brow[NT];
#pragma unroll
    for (int t = 0; t < NT; t++)
        brow[t] = BT + (size_t)(t * 16 + lmod) * K + hi * 8;

    v8f acc[NT] = {};
    for (int k0 = 0; k0 < K; k0 += 32) {
        v8bf alo = *(const v8bf*)(arow + k0);
        v8bf ahi = *(const v8bf*)(arow + k0 + 16);
        v16bf a = __builtin_shufflevector(alo, ahi, 0, 1, 2, 3, 4, 5, 6, 7,
                                          8, 9, 10, 11, 12, 13, 14, 15);
#pragma unroll
        for (int t = 0; t < NT; t++) {
            v8bf blo = *(const v8bf*)(brow[t] + k0);
            v8bf bhi = *(const v8bf*)(brow[t] + k0 + 16);
            v16bf b = __builtin_shufflevector(blo, bhi, 0, 1, 2, 3, 4, 5, 6, 7,
                                              8, 9, 10, 11, 12, 13, 14, 15);
            acc[t] = __builtin_amdgcn_wmma_f32_16x16x32_bf16(
                false, a, false, b, (short)0, acc[t], false, false);
        }
    }
    int mrow = mt * 16 + hi * 8;
#pragma unroll
    for (int t = 0; t < NT; t++) {
        int n = t * 16 + lmod;
#pragma unroll
        for (int v = 0; v < 8; v++)
            outf[(size_t)(mrow + v) * Nc + n] = acc[t][v];
    }
}

// ---------------- GCN graph kernels ----------------

__global__ void count_deg(const int* __restrict__ ei, float* deg, int E) {
    int e = blockIdx.x * blockDim.x + threadIdx.x;
    if (e < E) atomicAdd(&deg[ei[E + e]], 1.f);   // dst = edge_index[1][e]
}

__global__ void deg_to_dinv(float* deg, int n) {
    int i = blockIdx.x * blockDim.x + threadIdx.x;
    if (i < n) deg[i] = rsqrtf(fmaxf(deg[i], 1e-12f));
}

template <int F>
__global__ void gcn_scatter(const int* __restrict__ ei, int E,
                            const float* __restrict__ dinv,
                            const float* __restrict__ xw, float* agg) {
    const int groups = F / 4;
    int t = blockIdx.x * blockDim.x + threadIdx.x;
    int e = t / groups;
    if (e >= E) return;
    int f = (t % groups) * 4;
    int s = ei[e], d = ei[E + e];
    float nm = dinv[s] * dinv[d];
    const float* xs = xw + (size_t)s * F + f;
    float* ad = agg + (size_t)d * F + f;
    atomicAdd(&ad[0], xs[0] * nm);
    atomicAdd(&ad[1], xs[1] * nm);
    atomicAdd(&ad[2], xs[2] * nm);
    atomicAdd(&ad[3], xs[3] * nm);
}

__global__ void gcn_finish(const float* __restrict__ xw, const float* __restrict__ agg,
                           const float* __restrict__ dinv, const float* __restrict__ bias,
                           int F, int total, int doElu,
                           __bf16* outb, float* outf) {
    int i = blockIdx.x * blockDim.x + threadIdx.x;
    if (i >= total) return;
    int n = i / F, f = i % F;
    float di = dinv[n];
    float v = agg[i] + xw[i] * di * di + bias[f];
    if (doElu) v = v > 0.f ? v : (expf(v) - 1.f);
    if (outb) outb[i] = (__bf16)v;
    if (outf) outf[i] = v;
}

// ---------------- edge MLP head ----------------
__global__ void edge_head(const int* __restrict__ ei, int E,
                          const int* __restrict__ mask, int nsel,
                          const float* __restrict__ z,
                          const float* __restrict__ lw1, const float* __restrict__ lb1,
                          const float* __restrict__ lw2, const float* __restrict__ lb2,
                          float* __restrict__ out) {
    int i = blockIdx.x * blockDim.x + threadIdx.x;
    if (i >= nsel) return;
    int me = mask[i];
    int s = ei[me], d = ei[E + me];
    float e[32];
#pragma unroll
    for (int k = 0; k < 16; k++) e[k] = z[(size_t)s * 16 + k];
#pragma unroll
    for (int k = 0; k < 16; k++) e[16 + k] = z[(size_t)d * 16 + k];
    float h[16];
#pragma unroll
    for (int j = 0; j < 16; j++) {
        float a = lb1[j];
#pragma unroll
        for (int k = 0; k < 32; k++) a += e[k] * lw1[k * 16 + j];
        h[j] = a > 0.f ? a : (expf(a) - 1.f);
    }
    float o0 = lb2[0], o1 = lb2[1];
#pragma unroll
    for (int j = 0; j < 16; j++) { o0 += h[j] * lw2[j * 2]; o1 += h[j] * lw2[j * 2 + 1]; }
    out[2 * i]     = o0;
    out[2 * i + 1] = o1;
}

// ---------------- host-side orchestration ----------------

extern "C" void kernel_launch(void* const* d_in, const int* in_sizes, int n_in,
                              void* d_out, int out_size, void* d_ws, size_t ws_size,
                              hipStream_t stream) {
    const float* x    = (const float*)d_in[0];
    const int*   ei   = (const int*)d_in[1];
    const int*   mask = (const int*)d_in[2];
    const float* ln_g = (const float*)d_in[3];
    const float* ln_b = (const float*)d_in[4];
    const float* w0   = (const float*)d_in[5];  const float* b0  = (const float*)d_in[6];
    const float* w1   = (const float*)d_in[7];  const float* b1  = (const float*)d_in[8];
    const float* w2   = (const float*)d_in[9];  const float* b2  = (const float*)d_in[10];
    const float* c1w  = (const float*)d_in[11]; const float* c1b = (const float*)d_in[12];
    const float* c2w  = (const float*)d_in[13]; const float* c2b = (const float*)d_in[14];
    const float* lw1  = (const float*)d_in[15]; const float* lb1 = (const float*)d_in[16];
    const float* lw2  = (const float*)d_in[17]; const float* lb2 = (const float*)d_in[18];
    float* out = (float*)d_out;

    const int Nn   = NN;
    const int Ee   = in_sizes[1] / 2;
    const int nsel = in_sizes[2];

    char* ws = (char*)d_ws;
    size_t off = 0;
    auto alloc = [&](size_t bytes) {
        void* p = ws + off;
        off = (off + bytes + 255) & ~(size_t)255;
        return p;
    };
    __bf16* hB   = (__bf16*)alloc((size_t)Nn * 1024 * 2);
    __bf16* h0B  = (__bf16*)alloc((size_t)Nn * 512 * 2);
    __bf16* h1B  = (__bf16*)alloc((size_t)Nn * 128 * 2);
    __bf16* h2B  = (__bf16*)alloc((size_t)Nn * 64 * 2);
    float*  xw1  = (float*)alloc((size_t)Nn * 32 * 4);
    float*  agg1 = (float*)alloc((size_t)Nn * 32 * 4);
    __bf16* g1B  = (__bf16*)alloc((size_t)Nn * 32 * 2);
    float*  xw2  = (float*)alloc((size_t)Nn * 16 * 4);
    float*  agg2 = (float*)alloc((size_t)Nn * 16 * 4);
    float*  zf   = (float*)alloc((size_t)Nn * 16 * 4);
    float*  deg  = (float*)alloc((size_t)Nn * 4);
    __bf16* wb0  = (__bf16*)alloc((size_t)1024 * 512 * 2);   // transposed: 512 x 1024
    __bf16* wb1  = (__bf16*)alloc((size_t)512 * 128 * 2);    // 128 x 512
    __bf16* wb2  = (__bf16*)alloc((size_t)128 * 64 * 2);     // 64 x 128
    __bf16* wc1  = (__bf16*)alloc((size_t)64 * 32 * 2);      // 32 x 64
    __bf16* wc2  = (__bf16*)alloc((size_t)32 * 16 * 2);      // 16 x 32
    (void)ws_size; (void)n_in; (void)out_size;

    const int T = 256;
    auto nb = [](long n, int t) { return (int)((n + t - 1) / t); };

    // weights -> bf16 transposed (Nc x K)
    f32_to_bf16_T<<<nb(1024 * 512, T), T, 0, stream>>>(w0, wb0, 1024, 512);
    f32_to_bf16_T<<<nb(512 * 128, T), T, 0, stream>>>(w1, wb1, 512, 128);
    f32_to_bf16_T<<<nb(128 * 64, T), T, 0, stream>>>(w2, wb2, 128, 64);
    f32_to_bf16_T<<<nb(64 * 32, T), T, 0, stream>>>(c1w, wc1, 64, 32);
    f32_to_bf16_T<<<nb(32 * 16, T), T, 0, stream>>>(c2w, wc2, 32, 16);

    // pool + layernorm
    pool_ln<<<Nn, 256, 0, stream>>>(x, ln_g, ln_b, hB);

    // dense MLP: LDS-staged async-DMA WMMA GEMMs (K,Nc multiples of 64)
    int mblocks = (Nn / 16 + 7) / 8;   // 8 waves / block, 16 rows / wave
    gemm_bf16_lds<<<dim3(mblocks, 512 / 64), 256, 0, stream>>>(
        hB, wb0, b0, h0B, Nn, 1024, 512);
    gemm_bf16_lds<<<dim3(mblocks, 128 / 64), 256, 0, stream>>>(
        h0B, wb1, b1, h1B, Nn, 512, 128);
    gemm_bf16_lds<<<dim3(mblocks, 1), 256, 0, stream>>>(
        h1B, wb2, b2, h2B, Nn, 128, 64);

    // degree -> dinv  (self-loop contributes the initial 1.0)
    fill_f32<<<nb(Nn, T), T, 0, stream>>>(deg, 1.0f, Nn);
    count_deg<<<nb(Ee, T), T, 0, stream>>>(ei, deg, Ee);
    deg_to_dinv<<<nb(Nn, T), T, 0, stream>>>(deg, Nn);

    // conv1: xw1 = h2 @ c1w (WMMA direct), scatter, +self-loop +bias, ELU -> bf16
    gemm_bf16_small<2><<<dim3(mblocks, 1), 256, 0, stream>>>(h2B, wc1, xw1, Nn, 64, 32);
    fill_f32<<<nb((long)Nn * 32, T), T, 0, stream>>>(agg1, 0.f, Nn * 32);
    gcn_scatter<32><<<nb((long)Ee * 8, T), T, 0, stream>>>(ei, Ee, deg, xw1, agg1);
    gcn_finish<<<nb((long)Nn * 32, T), T, 0, stream>>>(
        xw1, agg1, deg, c1b, 32, Nn * 32, 1, g1B, nullptr);

    // conv2: xw2 = g1 @ c2w (WMMA direct), scatter, +self-loop +bias -> z (f32)
    gemm_bf16_small<1><<<dim3(mblocks, 1), 256, 0, stream>>>(g1B, wc2, xw2, Nn, 32, 16);
    fill_f32<<<nb((long)Nn * 16, T), T, 0, stream>>>(agg2, 0.f, Nn * 16);
    gcn_scatter<16><<<nb((long)Ee * 4, T), T, 0, stream>>>(ei, Ee, deg, xw2, agg2);
    gcn_finish<<<nb((long)Nn * 16, T), T, 0, stream>>>(
        xw2, agg2, deg, c2b, 16, Nn * 16, 0, nullptr, zf);

    // edge MLP head -> out (nsel x 2)
    edge_head<<<nb(nsel, T), T, 0, stream>>>(ei, Ee, mask, nsel, zf,
                                             lw1, lb1, lw2, lb2, out);
}
